// SegFormerXAttention_2173253451998
// MI455X (gfx1250) — matrix operand order repaired and checked by reading, not compiled, not run on hardware
//
#include <hip/hip_runtime.h>
#include <hip/hip_bf16.h>
#include <cstdint>
#include <cstddef>

// ---------------------------------------------------------------------------
// CDNA5 (gfx1250) bf16 WMMA implementation of SegFormer cross-attention.
// wave32; V_WMMA_F32_16X16X32_BF16 everywhere; f32 accumulate; flash softmax.
// GEMM: 32x32 per wave (4 WMMA / 8 ds_load_b128 per K-step), BM=128 BN=64.
// ---------------------------------------------------------------------------

typedef __attribute__((ext_vector_type(16))) __bf16 v16bf;
typedef __attribute__((ext_vector_type(8)))  __bf16 v8bf;
typedef __attribute__((ext_vector_type(8)))  float  v8f;

#define DMODEL 1024
#define NHEAD  16
#define DHEAD  64

static __device__ __forceinline__ int lane_id() { return (int)(threadIdx.x & 31); }

// A fragment (16x32, MxK) from a row-major source (ld in elements).
// ISA layout: lane l: M = l%16, half h = l/16; elems 0..7 -> K = h*8 + i,
// elems 8..15 -> K = 16 + h*8 + (i-8). Two contiguous 16B chunks.
static __device__ __forceinline__ v16bf load_frag_a(const __bf16* base, int ld) {
  int l = lane_id();
  int m = l & 15, hf = l >> 4;
  const __bf16* p = base + (size_t)m * ld + hf * 8;
  union { v16bf v; v8bf h[2]; } u;
  u.h[0] = *(const v8bf*)(p);
  u.h[1] = *(const v8bf*)(p + 16);
  return u.v;
}

// B fragment (32x16, KxN) where source is stored N-major:
// row n of the source = output column n, contiguous K elements (ld in elems).
// ISA layout: lane l: N = l%16, h = l/16; elems 0..7 -> K = h*8 + i,
// elems 8..15 -> K = 16 + h*8 + (i-8).
static __device__ __forceinline__ v16bf load_frag_b(const __bf16* base, int ld) {
  int l = lane_id();
  int n = l & 15, kh = (l >> 4) * 8;
  const __bf16* p = base + (size_t)n * ld + kh;
  union { v16bf v; v8bf h[2]; } u;
  u.h[0] = *(const v8bf*)(p);
  u.h[1] = *(const v8bf*)(p + 16);
  return u.v;
}

static __device__ __forceinline__ v8f wmma_bf16(v16bf a, v16bf b, v8f c) {
  return __builtin_amdgcn_wmma_f32_16x16x32_bf16(false, a, false, b,
                                                 (short)0, c, false, false);
}

// ---------------------------------------------------------------------------
// f32 -> bf16 conversion (grid-stride)
// ---------------------------------------------------------------------------
__global__ void cvt_f32_bf16(const float* __restrict__ in,
                             __bf16* __restrict__ out, int n) {
  int i = blockIdx.x * blockDim.x + threadIdx.x;
  int stride = gridDim.x * blockDim.x;
  for (; i < n; i += stride) out[i] = (__bf16)in[i];
}

// ---------------------------------------------------------------------------
// Tiled bf16 WMMA GEMM:  C[M,N] = A[M,K] @ W[K,N] + bias
//   BM=128, BN=64, BK=32, 256 threads (8 waves in 4x2).
//   Each wave: 32x32 output = 2 A frags x 2 B frags -> 4 WMMA per K-step.
// Epilogue modes:
//   0: bf16 row-major, with batch row remap:
//        row_out = (m/Lin)*Lout + row0 + (m%Lin);  outb[row_out*ldo + n]
//   1: bf16 transposed (value matrices stored [B, N, Lout]):
//        outb[((b*N + n)*ldo) + row0 + (m%Lin)]   (8 contiguous M -> 1 b128)
//   2: f32 + bias + residual:  outf[m*N + n] = acc + bias[n] + resid[m*N + n]
// ---------------------------------------------------------------------------
__global__ __launch_bounds__(256) void gemm_bf16_kernel(
    const __bf16* __restrict__ A, int lda,
    const __bf16* __restrict__ W, int ldw,
    const float* __restrict__ bias,
    int M, int N, int K,
    int Lin, int Lout, int row0,
    int mode,
    __bf16* __restrict__ outb, int ldo,
    const float* __restrict__ resid,
    float* __restrict__ outf) {
  __shared__ __bf16 As[128 * 40];  // row-major, padded stride 40 halves
  __shared__ __bf16 Wt[64 * 40];   // transposed: Wt[n][k]
  const int tid = (int)threadIdx.x;
  const int l = tid & 31;
  const int w = tid >> 5;
  const int m0 = blockIdx.y * 128;
  const int n0 = blockIdx.x * 64;
  const int wm = w >> 1;           // 0..3 -> 32-row band
  const int wn = w & 1;            // 0..1 -> 32-col band

  v8f acc00 = {}, acc01 = {}, acc10 = {}, acc11 = {};

  // staging assignments (constant across K loop)
  const int ar = tid >> 1;               // A row 0..127
  const int ac = (tid & 1) * 16;         // A col 0 or 16
  const int wr = tid >> 3;               // W k-row 0..31
  const int wc = (tid & 7) * 8;          // W n-col 0,8,..,56

  for (int k0 = 0; k0 < K; k0 += 32) {
    {
      const __bf16* src = A + (size_t)(m0 + ar) * lda + k0 + ac;
      v8bf a0 = *(const v8bf*)(src);
      v8bf a1 = *(const v8bf*)(src + 8);
      *(v8bf*)(&As[ar * 40 + ac]) = a0;
      *(v8bf*)(&As[ar * 40 + ac + 8]) = a1;
    }
    {
      const __bf16* src = W + (size_t)(k0 + wr) * ldw + n0 + wc;
      v8bf wv = *(const v8bf*)(src);
#pragma unroll
      for (int j = 0; j < 8; ++j) Wt[(wc + j) * 40 + wr] = wv[j];
    }
    if (k0 + 32 < K) {
      // prefetch next K-slice (lowers to global_prefetch_b8)
      __builtin_prefetch(A + (size_t)(m0 + ar) * lda + k0 + 32 + ac, 0, 1);
      __builtin_prefetch(W + (size_t)(k0 + 32 + wr) * ldw + n0 + wc, 0, 1);
    }
    __syncthreads();
    v16bf a0 = load_frag_a(&As[(wm * 32) * 40], 40);
    v16bf a1 = load_frag_a(&As[(wm * 32 + 16) * 40], 40);
    v16bf b0 = load_frag_b(&Wt[(wn * 32) * 40], 40);
    v16bf b1 = load_frag_b(&Wt[(wn * 32 + 16) * 40], 40);
    acc00 = wmma_bf16(a0, b0, acc00);
    acc01 = wmma_bf16(a0, b1, acc01);
    acc10 = wmma_bf16(a1, b0, acc10);
    acc11 = wmma_bf16(a1, b1, acc11);
    __syncthreads();
  }

#pragma unroll
  for (int mi = 0; mi < 2; ++mi) {
#pragma unroll
    for (int ni = 0; ni < 2; ++ni) {
      v8f acc = (mi == 0) ? (ni == 0 ? acc00 : acc01)
                          : (ni == 0 ? acc10 : acc11);
      const int mbase = m0 + wm * 32 + mi * 16 + (l >> 4) * 8;
      const int nglob = n0 + wn * 32 + ni * 16 + (l & 15);
      float bia = bias[nglob];
      if (mode == 0) {
#pragma unroll
        for (int r = 0; r < 8; ++r) {
          int m = mbase + r;
          int bidx = m / Lin, rr = m - bidx * Lin;
          size_t row = (size_t)bidx * Lout + row0 + rr;
          outb[row * (size_t)ldo + nglob] = (__bf16)(acc[r] + bia);
        }
      } else if (mode == 1) {
        int bidx = mbase / Lin, rr = mbase - bidx * Lin;
        size_t idx = ((size_t)bidx * N + nglob) * (size_t)ldo + row0 + rr;
        v8bf pk;
#pragma unroll
        for (int r = 0; r < 8; ++r) pk[r] = (__bf16)(acc[r] + bia);
        *(v8bf*)(&outb[idx]) = pk;
      } else {
#pragma unroll
        for (int r = 0; r < 8; ++r) {
          size_t idx = (size_t)(mbase + r) * N + nglob;
          outf[idx] = acc[r] + bia + resid[idx];
        }
      }
    }
  }
}

// ---------------------------------------------------------------------------
// Flash attention over concatenated keys [vid(LV); usr(LT)].
// One wave = one (b, h, 16-query tile). Streams 32 keys / iter.
// Query projection switches at the LV boundary (qA vs qB), matching the
// reference's concat of two differently-projected logit blocks.
// Masked logit = -10000 * scale = -1250 (mask applied BEFORE scale).
// ---------------------------------------------------------------------------
__global__ __launch_bounds__(256) void flash_attn_kernel(
    const __bf16* __restrict__ qA, const __bf16* __restrict__ qB,
    const __bf16* __restrict__ Kc,   // [B, Lcat, D]
    const __bf16* __restrict__ Vt,   // [B, D, Lcat]  (transposed values)
    const unsigned char* __restrict__ mQ,
    const unsigned char* __restrict__ mKv,
    const unsigned char* __restrict__ mKu,
    __bf16* __restrict__ outb,       // [B, Lq, D]
    int Lq, int Lcat, int LV_, int LT_) {
  __shared__ __bf16 Pall[8 * 16 * 40];     // per-wave P transpose staging
  const int l = (int)(threadIdx.x & 31);
  const int w = (int)(threadIdx.x >> 5);
  __bf16* P = &Pall[w * 16 * 40];

  const int nQT = Lq >> 4;
  int tile = blockIdx.x * 8 + w;
  int b = tile / (NHEAD * nQT);
  int rem = tile - b * (NHEAD * nQT);
  int h = rem / nQT;
  int qt = rem - h * nQT;

  const __bf16* qAb = qA + ((size_t)(b * Lq + qt * 16)) * DMODEL + h * DHEAD;
  const __bf16* qBb = qB + ((size_t)(b * Lq + qt * 16)) * DMODEL + h * DHEAD;
  v16bf qa0 = load_frag_a(qAb, DMODEL);        // d 0..31
  v16bf qa1 = load_frag_a(qAb + 32, DMODEL);   // d 32..63
  v16bf qb0 = load_frag_a(qBb, DMODEL);
  v16bf qb1 = load_frag_a(qBb + 32, DMODEL);

  const int nl = l & 15;
  const int mrow = (l >> 4) * 8;   // this lane's 8 query rows: mrow..mrow+7
  unsigned qm = 0;
  {
    const unsigned char* mq = mQ + (size_t)b * Lq + qt * 16 + mrow;
#pragma unroll
    for (int j = 0; j < 8; ++j) qm |= (mq[j] ? 1u : 0u) << j;
  }

  float rmax[8], rsum[8];
  v8f o0 = {}, o1 = {}, o2 = {}, o3 = {};
#pragma unroll
  for (int r = 0; r < 8; ++r) { rmax[r] = -3.0e38f; rsum[r] = 0.0f; }

  const int nKT = Lcat >> 5;
  for (int t = 0; t < nKT; ++t) {
    const int kb = t * 32;
    const bool seg0 = (kb < LV_);
    v16bf a0, a1;
    if (seg0) { a0 = qa0; a1 = qa1; } else { a0 = qb0; a1 = qb1; }
    const unsigned char* mk = seg0 ? (mKv + (size_t)b * LV_ + kb)
                                   : (mKu + (size_t)b * LT_ + (kb - LV_));
    const bool km0 = mk[nl] != 0;
    const bool km1 = mk[16 + nl] != 0;

    // S = Q @ K^T  (two 16-key n-tiles, dh=64 => 2 k-steps each)
    const __bf16* kbase = Kc + ((size_t)(b * Lcat + kb)) * DMODEL + h * DHEAD;
    const __bf16* vbase = Vt + ((size_t)b * DMODEL + h * DHEAD) * Lcat + kb;
    if (t + 1 < nKT) {
      __builtin_prefetch(kbase + (size_t)32 * DMODEL, 0, 1);
      __builtin_prefetch(vbase + 32, 0, 1);
    }
    v8f s0 = {}, s1 = {};
    s0 = wmma_bf16(a0, load_frag_b(kbase, DMODEL), s0);
    s0 = wmma_bf16(a1, load_frag_b(kbase + 32, DMODEL), s0);
    s1 = wmma_bf16(a0, load_frag_b(kbase + (size_t)16 * DMODEL, DMODEL), s1);
    s1 = wmma_bf16(a1, load_frag_b(kbase + (size_t)16 * DMODEL + 32, DMODEL), s1);

    // mask (before scale, as in reference) + scale
    float nm[8];
#pragma unroll
    for (int r = 0; r < 8; ++r) {
      bool qr = (qm >> r) & 1u;
      float v0 = (qr && km0) ? s0[r] * 0.125f : -1250.0f;
      float v1 = (qr && km1) ? s1[r] * 0.125f : -1250.0f;
      s0[r] = v0; s1[r] = v1;
      nm[r] = fmaxf(v0, v1);
    }
    // row max across the 16 lanes holding this row (xor<16 stays in half)
#pragma unroll
    for (int ofs = 1; ofs < 16; ofs <<= 1) {
#pragma unroll
      for (int r = 0; r < 8; ++r) nm[r] = fmaxf(nm[r], __shfl_xor(nm[r], ofs, 32));
    }
    float corr[8], psum[8];
#pragma unroll
    for (int r = 0; r < 8; ++r) {
      float mnew = fmaxf(rmax[r], nm[r]);
      corr[r] = __expf(rmax[r] - mnew);
      rmax[r] = mnew;
      float p0 = __expf(s0[r] - mnew);
      float p1 = __expf(s1[r] - mnew);
      s0[r] = p0; s1[r] = p1;
      psum[r] = p0 + p1;
    }
#pragma unroll
    for (int ofs = 1; ofs < 16; ofs <<= 1) {
#pragma unroll
      for (int r = 0; r < 8; ++r) psum[r] += __shfl_xor(psum[r], ofs, 32);
    }
#pragma unroll
    for (int r = 0; r < 8; ++r) rsum[r] = rsum[r] * corr[r] + psum[r];

    // transpose P (C-layout -> A-layout) through wave-private LDS
#pragma unroll
    for (int r = 0; r < 8; ++r) {
      P[(mrow + r) * 40 + nl]      = (__bf16)s0[r];
      P[(mrow + r) * 40 + 16 + nl] = (__bf16)s1[r];
    }
    v16bf pa = load_frag_a(P, 40);

    // rescale running O (row stats align with C-fragment rows per lane)
#pragma unroll
    for (int r = 0; r < 8; ++r) {
      o0[r] *= corr[r]; o1[r] *= corr[r]; o2[r] *= corr[r]; o3[r] *= corr[r];
    }

    // O += P @ V   (V stored transposed: contiguous keys per d-row)
    o0 = wmma_bf16(pa, load_frag_b(vbase, Lcat), o0);
    o1 = wmma_bf16(pa, load_frag_b(vbase + (size_t)16 * Lcat, Lcat), o1);
    o2 = wmma_bf16(pa, load_frag_b(vbase + (size_t)32 * Lcat, Lcat), o2);
    o3 = wmma_bf16(pa, load_frag_b(vbase + (size_t)48 * Lcat, Lcat), o3);
  }

  // epilogue: normalize and store bf16 row-major for the output projection
  __bf16* ob = outb + ((size_t)(b * Lq + qt * 16)) * DMODEL + h * DHEAD;
#pragma unroll
  for (int r = 0; r < 8; ++r) {
    float inv = 1.0f / rsum[r];
    size_t rowo = (size_t)(mrow + r) * DMODEL;
    ob[rowo + nl]      = (__bf16)(o0[r] * inv);
    ob[rowo + 16 + nl] = (__bf16)(o1[r] * inv);
    ob[rowo + 32 + nl] = (__bf16)(o2[r] * inv);
    ob[rowo + 48 + nl] = (__bf16)(o3[r] * inv);
  }
}

// ---------------------------------------------------------------------------
// Row LayerNorm (eps = 1e-12), one block per row of D=1024.
// ---------------------------------------------------------------------------
__global__ __launch_bounds__(256) void layernorm_kernel(
    const float* __restrict__ x, const float* __restrict__ g,
    const float* __restrict__ bta, float* __restrict__ out) {
  const int row = blockIdx.x;
  const float* xr = x + (size_t)row * DMODEL;
  float s = 0.f, ss = 0.f;
  for (int i = (int)threadIdx.x; i < DMODEL; i += 256) {
    float v = xr[i]; s += v; ss += v * v;
  }
#pragma unroll
  for (int ofs = 1; ofs < 32; ofs <<= 1) {
    s += __shfl_xor(s, ofs, 32);
    ss += __shfl_xor(ss, ofs, 32);
  }
  __shared__ float sh[16];
  int w = (int)(threadIdx.x >> 5), l = (int)(threadIdx.x & 31);
  if (l == 0) { sh[w] = s; sh[8 + w] = ss; }
  __syncthreads();
  if (threadIdx.x == 0) {
    float ts = 0.f, tss = 0.f;
#pragma unroll
    for (int i = 0; i < 8; ++i) { ts += sh[i]; tss += sh[8 + i]; }
    sh[0] = ts; sh[8] = tss;
  }
  __syncthreads();
  float mu = sh[0] * (1.0f / DMODEL);
  float var = sh[8] * (1.0f / DMODEL) - mu * mu;
  float rstd = rsqrtf(var + 1e-12f);
  for (int i = (int)threadIdx.x; i < DMODEL; i += 256) {
    out[(size_t)row * DMODEL + i] = (xr[i] - mu) * rstd * g[i] + bta[i];
  }
}

// ---------------------------------------------------------------------------
// Host launcher
// ---------------------------------------------------------------------------
extern "C" void kernel_launch(void* const* d_in, const int* in_sizes, int n_in,
                              void* d_out, int out_size, void* d_ws, size_t ws_size,
                              hipStream_t stream) {
  (void)in_sizes; (void)n_in; (void)out_size; (void)ws_size;
  const int B_ = 2, LV_ = 1024, LT_ = 512, D_ = DMODEL;
  const int Lcat = LV_ + LT_;
  const int DD = D_ * D_;
  const int Mv = B_ * LV_;   // 2048
  const int Mu = B_ * LT_;   // 1024

  const float* vid_feat = (const float*)d_in[0];
  const float* usr_feat = (const float*)d_in[1];
  const unsigned char* vid_mask = (const unsigned char*)d_in[2];
  const unsigned char* usr_mask = (const unsigned char*)d_in[3];
  const float* v2v_W = (const float*)d_in[4];
  const float* v2v_b = (const float*)d_in[5];
  const float* t2v_W = (const float*)d_in[6];
  const float* t2v_b = (const float*)d_in[7];
  const float* v2t_W = (const float*)d_in[8];
  const float* v2t_b = (const float*)d_in[9];
  const float* t2t_W = (const float*)d_in[10];
  const float* t2t_b = (const float*)d_in[11];
  const float* ffu_W = (const float*)d_in[12];
  const float* ffu_b = (const float*)d_in[13];
  const float* ffv_W = (const float*)d_in[14];
  const float* ffv_b = (const float*)d_in[15];
  const float* lnu_g = (const float*)d_in[16];
  const float* lnu_b = (const float*)d_in[17];
  const float* lnv_g = (const float*)d_in[18];
  const float* lnv_b = (const float*)d_in[19];

  char* p = (char*)d_ws;
  auto take = [&](size_t elems, size_t esz) -> void* {
    void* q = (void*)p;
    p += ((elems * esz + 255) & ~(size_t)255);
    return q;
  };
  __bf16* vidb   = (__bf16*)take((size_t)Mv * D_, 2);
  __bf16* usrb   = (__bf16*)take((size_t)Mu * D_, 2);
  __bf16* Wv2v   = (__bf16*)take((size_t)3 * DD, 2);
  __bf16* Wt2v   = (__bf16*)take((size_t)3 * DD, 2);
  __bf16* Wv2t   = (__bf16*)take((size_t)3 * DD, 2);
  __bf16* Wt2t   = (__bf16*)take((size_t)3 * DD, 2);
  __bf16* ffub   = (__bf16*)take((size_t)DD, 2);
  __bf16* ffvb   = (__bf16*)take((size_t)DD, 2);
  __bf16* q_v2v  = (__bf16*)take((size_t)Mv * D_, 2);
  __bf16* q_t2v  = (__bf16*)take((size_t)Mv * D_, 2);
  __bf16* q_v2t  = (__bf16*)take((size_t)Mu * D_, 2);
  __bf16* q_t2t  = (__bf16*)take((size_t)Mu * D_, 2);
  __bf16* kcat_v = (__bf16*)take((size_t)B_ * Lcat * D_, 2);
  __bf16* kcat_t = (__bf16*)take((size_t)B_ * Lcat * D_, 2);
  __bf16* vt_v   = (__bf16*)take((size_t)B_ * D_ * Lcat, 2);
  __bf16* vt_t   = (__bf16*)take((size_t)B_ * D_ * Lcat, 2);
  __bf16* attn_v = (__bf16*)take((size_t)Mv * D_, 2);
  __bf16* attn_t = (__bf16*)take((size_t)Mu * D_, 2);
  float*  tmp_v  = (float*)take((size_t)Mv * D_, 4);
  float*  tmp_t  = (float*)take((size_t)Mu * D_, 4);

  auto cvt = [&](const float* in, __bf16* out, int n) {
    cvt_f32_bf16<<<256, 256, 0, stream>>>(in, out, n);
  };
  cvt(vid_feat, vidb, Mv * D_);
  cvt(usr_feat, usrb, Mu * D_);
  cvt(v2v_W, Wv2v, 3 * DD);
  cvt(t2v_W, Wt2v, 3 * DD);
  cvt(v2t_W, Wv2t, 3 * DD);
  cvt(t2t_W, Wt2t, 3 * DD);
  cvt(ffu_W, ffub, DD);
  cvt(ffv_W, ffvb, DD);

  auto gemm = [&](const __bf16* A, int M, int Lin,
                  const __bf16* W, const float* bias,
                  int mode, __bf16* ob, int Lout, int row0, int ldo,
                  const float* resid, float* of) {
    dim3 grid(D_ / 64, M / 128);
    gemm_bf16_kernel<<<grid, 256, 0, stream>>>(A, D_, W, D_, bias, M, D_, D_,
                                               Lin, Lout, row0, mode,
                                               ob, ldo, resid, of);
  };

  // Q projections (row-major bf16)
  gemm(vidb, Mv, LV_, Wv2v + 0 * DD, v2v_b + 0 * D_, 0, q_v2v, LV_, 0, D_, nullptr, nullptr);
  gemm(vidb, Mv, LV_, Wt2v + 0 * DD, t2v_b + 0 * D_, 0, q_t2v, LV_, 0, D_, nullptr, nullptr);
  gemm(usrb, Mu, LT_, Wv2t + 0 * DD, v2t_b + 0 * D_, 0, q_v2t, LT_, 0, D_, nullptr, nullptr);
  gemm(usrb, Mu, LT_, Wt2t + 0 * DD, t2t_b + 0 * D_, 0, q_t2t, LT_, 0, D_, nullptr, nullptr);
  // K projections, concatenated [vid; usr] per side
  gemm(vidb, Mv, LV_, Wv2v + 1 * DD, v2v_b + 1 * D_, 0, kcat_v, Lcat, 0,   D_, nullptr, nullptr);
  gemm(usrb, Mu, LT_, Wt2v + 1 * DD, t2v_b + 1 * D_, 0, kcat_v, Lcat, LV_, D_, nullptr, nullptr);
  gemm(vidb, Mv, LV_, Wv2t + 1 * DD, v2t_b + 1 * D_, 0, kcat_t, Lcat, 0,   D_, nullptr, nullptr);
  gemm(usrb, Mu, LT_, Wt2t + 1 * DD, t2t_b + 1 * D_, 0, kcat_t, Lcat, LV_, D_, nullptr, nullptr);
  // V projections, stored transposed [B, D, Lcat]
  gemm(vidb, Mv, LV_, Wv2v + 2 * DD, v2v_b + 2 * D_, 1, vt_v, 0, 0,   Lcat, nullptr, nullptr);
  gemm(usrb, Mu, LT_, Wt2v + 2 * DD, t2v_b + 2 * D_, 1, vt_v, 0, LV_, Lcat, nullptr, nullptr);
  gemm(vidb, Mv, LV_, Wv2t + 2 * DD, v2t_b + 2 * D_, 1, vt_t, 0, 0,   Lcat, nullptr, nullptr);
  gemm(usrb, Mu, LT_, Wt2t + 2 * DD, t2t_b + 2 * D_, 1, vt_t, 0, LV_, Lcat, nullptr, nullptr);

  // Flash attention, vid queries then usr queries
  {
    int blocksV = B_ * NHEAD * (LV_ / 16) / 8;   // 256
    flash_attn_kernel<<<blocksV, 256, 0, stream>>>(
        q_v2v, q_t2v, kcat_v, vt_v, vid_mask, vid_mask, usr_mask,
        attn_v, LV_, Lcat, LV_, LT_);
    int blocksT = B_ * NHEAD * (LT_ / 16) / 8;   // 128
    flash_attn_kernel<<<blocksT, 256, 0, stream>>>(
        q_v2t, q_t2t, kcat_t, vt_t, usr_mask, vid_mask, usr_mask,
        attn_t, LT_, Lcat, LV_, LT_);
  }

  // Output projection + residual (f32), then LayerNorm into d_out
  gemm(attn_v, Mv, LV_, ffvb, ffv_b, 2, nullptr, LV_, 0, D_, vid_feat, tmp_v);
  gemm(attn_t, Mu, LT_, ffub, ffu_b, 2, nullptr, LT_, 0, D_, usr_feat, tmp_t);

  float* out_vid = (float*)d_out;                       // [B, LV, D]
  float* out_usr = out_vid + (size_t)Mv * D_;           // [B, LT, D]
  layernorm_kernel<<<Mv, 256, 0, stream>>>(tmp_v, lnv_g, lnv_b, out_vid);
  layernorm_kernel<<<Mu, 256, 0, stream>>>(tmp_t, lnu_g, lnu_b, out_usr);
}
